// IJGNN3_43920335569131
// MI455X (gfx1250) — matrix operand order repaired and checked by reading, not compiled, not run on hardware
//
#include <hip/hip_runtime.h>
#include <hip/hip_bf16.h>
#include <math.h>

// ---------------------------------------------------------------------------
// CDNA5 (gfx1250) GRU-MPNN. Wave32, v_wmma_f32_16x16x32_bf16 for all GEMMs,
// f32 accumulate, fused edge-GRU (two GEMMs + gates + scatter-add) per tile.
// A-operand staging uses GLOBAL_LOAD_ASYNC_TO_LDS_B128 (ASYNCcnt) when the
// toolchain exposes the builtin; falls back to VGPR staging otherwise.
// ---------------------------------------------------------------------------

typedef __attribute__((ext_vector_type(16))) __bf16 bf16x16;
typedef __attribute__((ext_vector_type(8)))  __bf16 bf16x8;
typedef __attribute__((ext_vector_type(8)))  float  f32x8;
typedef __attribute__((ext_vector_type(4)))  float  f32x4;
typedef __attribute__((ext_vector_type(4)))  int    v4i;

#define NNODE 10000
#define NEDGE 320000
#define HID   256

#if defined(__gfx1250__) && __has_builtin(__builtin_amdgcn_global_load_async_to_lds_b128)
#define HAVE_ASYNC_LDS 1
#else
#define HAVE_ASYNC_LDS 0
#endif

#if HAVE_ASYNC_LDS
typedef __attribute__((address_space(1))) v4i* gv4i_p;  // global (HIP __device__)
typedef __attribute__((address_space(3))) v4i* lv4i_p;  // LDS
__device__ __forceinline__ void async_cp16(const void* gsrc, void* ldst) {
  __builtin_amdgcn_global_load_async_to_lds_b128((gv4i_p)gsrc, (lv4i_p)ldst, 0, 0);
}
__device__ __forceinline__ void async_wait0() {
#if __has_builtin(__builtin_amdgcn_s_wait_asynccnt)
  __builtin_amdgcn_s_wait_asynccnt(0);
#else
  asm volatile("s_wait_asynccnt 0x0" ::: "memory");
#endif
}
#endif

__device__ __forceinline__ f32x8 wmma_bf16(bf16x16 a, bf16x16 b, f32x8 c) {
  // v_wmma_f32_16x16x32_bf16  D = A(16x32) * B(32x16) + C
  return __builtin_amdgcn_wmma_f32_16x16x32_bf16(
      /*neg_a=*/false, a, /*neg_b=*/false, b,
      /*c_mod=*/(short)0, c, /*reuse_a=*/false, /*reuse_b=*/false);
}

// B fragment (32x16, bf16): lane l holds col N=l&15,
// K = kk + kh*8 + [0..7] and kk + 16 + kh*8 + [0..7], kh = l>>4.  W is [Nout, K] row-major.
__device__ __forceinline__ bf16x16 load_b_frag(const __bf16* __restrict__ W,
                                               int ldw, int n0, int kk, int lane) {
  int n  = n0 + (lane & 15);
  int kh = (lane >> 4) * 8;
  const __bf16* p = W + (size_t)n * ldw + kk + kh;
  bf16x8 lo = *(const bf16x8*)p;
  bf16x8 hi = *(const bf16x8*)(p + 16);
  bf16x16 o;
#pragma unroll
  for (int i = 0; i < 8; ++i) { o[i] = lo[i]; o[8 + i] = hi[i]; }
  return o;
}

// A fragment (16x32, bf16) from LDS: lane l: row = r0 + (l&15);
// K = kk + kh*8 + [0..7] and kk + 16 + kh*8 + [0..7], kh = l>>4.
__device__ __forceinline__ bf16x16 load_a_frag_bf16(const __bf16* As, int ld,
                                                    int r0, int kk, int lane) {
  int row = r0 + (lane & 15);
  int kh  = (lane >> 4) * 8;
  const __bf16* p = As + row * ld + kk + kh;
  bf16x8 lo = *(const bf16x8*)p;        // 16B ds load
  bf16x8 hi = *(const bf16x8*)(p + 16); // 16B ds load
  bf16x16 o;
#pragma unroll
  for (int i = 0; i < 8; ++i) { o[i] = lo[i]; o[8 + i] = hi[i]; }
  return o;
}

// A fragment from f32 LDS, converted to bf16 on the fly.
__device__ __forceinline__ bf16x16 load_a_frag_f32(const float* As, int ld,
                                                   int r0, int kk, int lane) {
  int row = r0 + (lane & 15);
  int kh  = (lane >> 4) * 8;
  const f32x4* q = (const f32x4*)(As + row * ld + kk + kh);
  f32x4 a0 = q[0], a1 = q[1], a2 = q[4], a3 = q[5];
  bf16x16 o;
#pragma unroll
  for (int i = 0; i < 4; ++i) {
    o[i]      = (__bf16)a0[i];
    o[4 + i]  = (__bf16)a1[i];
    o[8 + i]  = (__bf16)a2[i];
    o[12 + i] = (__bf16)a3[i];
  }
  return o;
}

__device__ __forceinline__ float sigmoidf_(float x) {
  return 1.0f / (1.0f + __expf(-x));
}

// ---------------------------------------------------------------------------
// fused edge GRU: per 32-edge block:
//   gi = [h_src|h_dst] @ Wih.T  (K=512) ; gh = h_e @ Whh.T (K=256)
//   r/z accumulated as gi+gh directly in the WMMA accumulator; n parts kept
//   separate. Then gates + state update + atomic scatter into agg[dst].
// ---------------------------------------------------------------------------
__global__ __launch_bounds__(256, 1)
void edge_gru_kernel(const __bf16* __restrict__ unf_bf, float* __restrict__ uef,
                     const int* __restrict__ src, const int* __restrict__ dst,
                     const __bf16* __restrict__ Wih, const __bf16* __restrict__ Whh,
                     const float* __restrict__ bih, const float* __restrict__ bhh,
                     float* __restrict__ agg) {
  __shared__ __bf16 xe[32 * 512];  // gathered [h_src | h_dst] per edge (bf16)
  __shared__ float  hh[32 * 256];  // current edge hidden state (f32)
  __shared__ int    sd[64];        // src[0..31], dst[32..63]

  const int t  = threadIdx.x;
  const int e0 = blockIdx.x * 32;

  if (t < 32) { sd[t] = src[e0 + t]; sd[32 + t] = dst[e0 + t]; }
  __syncthreads();

#if HAVE_ASYNC_LDS
  {
    // xe gather: 64 contiguous 512B segments (edge x {src,dst}); 32 lanes x 16B
    const int lane = t & 31;
    for (int seg = t >> 5; seg < 64; seg += 8) {
      int e = seg >> 1;
      int node = (seg & 1) ? sd[32 + e] : sd[e];
      async_cp16(unf_bf + (size_t)node * 256 + lane * 8, xe + seg * 256 + lane * 8);
    }
    // h stage: contiguous 32KB
    const f32x4* up = (const f32x4*)(uef + (size_t)e0 * 256);
    for (int idx = t; idx < 2048; idx += 256) async_cp16(up + idx, (f32x4*)hh + idx);
    async_wait0();
  }
#else
  for (int idx = t; idx < 32 * 512; idx += 256) {
    int e = idx >> 9, k = idx & 511;
    int node = (k < 256) ? sd[e] : sd[32 + e];
    xe[idx] = unf_bf[(size_t)node * 256 + (k & 255)];
  }
  {
    const f32x4* up = (const f32x4*)(uef + (size_t)e0 * 256);
    f32x4* hp = (f32x4*)hh;
    for (int idx = t; idx < 2048; idx += 256) hp[idx] = up[idx];
  }
#endif
  __syncthreads();

  const int wave = t >> 5, lane = t & 31;

  for (int cb = wave; cb < 16; cb += 8) {   // 16 column blocks of 16
    const int c0 = cb * 16;
    f32x8 Sr[2] = {}, Sz[2] = {}, Gin[2] = {}, Ghn[2] = {};

    // gi over K=512 (16 k-steps), 3 column tiles, 2 row slabs -> 6 wmma/step
#pragma unroll 4
    for (int ks = 0; ks < 16; ++ks) {
      const int kk = ks * 32;
      bf16x16 br = load_b_frag(Wih, 512, c0,        kk, lane);
      bf16x16 bz = load_b_frag(Wih, 512, 256 + c0,  kk, lane);
      bf16x16 bn = load_b_frag(Wih, 512, 512 + c0,  kk, lane);
#pragma unroll
      for (int s = 0; s < 2; ++s) {
        bf16x16 a = load_a_frag_bf16(xe, 512, s * 16, kk, lane);
        Sr[s]  = wmma_bf16(a, br, Sr[s]);
        Sz[s]  = wmma_bf16(a, bz, Sz[s]);
        Gin[s] = wmma_bf16(a, bn, Gin[s]);
      }
    }
    // gh over K=256 (8 k-steps); r/z go into the same accumulators
#pragma unroll 4
    for (int ks = 0; ks < 8; ++ks) {
      const int kk = ks * 32;
      bf16x16 br = load_b_frag(Whh, 256, c0,        kk, lane);
      bf16x16 bz = load_b_frag(Whh, 256, 256 + c0,  kk, lane);
      bf16x16 bn = load_b_frag(Whh, 256, 512 + c0,  kk, lane);
#pragma unroll
      for (int s = 0; s < 2; ++s) {
        bf16x16 a = load_a_frag_f32(hh, 256, s * 16, kk, lane);
        Sr[s]  = wmma_bf16(a, br, Sr[s]);
        Sz[s]  = wmma_bf16(a, bz, Sz[s]);
        Ghn[s] = wmma_bf16(a, bn, Ghn[s]);
      }
    }

    const int   c    = c0 + (lane & 15);
    const float br_  = bih[c]       + bhh[c];
    const float bz_  = bih[256 + c] + bhh[256 + c];
    const float bin_ = bih[512 + c];
    const float bhn_ = bhh[512 + c];
    const int   rtop = (lane >> 4) * 8;

#pragma unroll
    for (int s = 0; s < 2; ++s)
#pragma unroll
      for (int r = 0; r < 8; ++r) {
        const int erow = s * 16 + rtop + r;
        float rg = sigmoidf_(Sr[s][r] + br_);
        float zg = sigmoidf_(Sz[s][r] + bz_);
        float ng = tanhf(Gin[s][r] + bin_ + rg * (Ghn[s][r] + bhn_));
        float h  = hh[erow * 256 + c];
        float hn = (1.0f - zg) * ng + zg * h;
        uef[(size_t)(e0 + erow) * 256 + c] = hn;
        atomicAdd(&agg[(size_t)sd[32 + erow] * 256 + c], hn);  // segment_sum
      }
  }
}

// ---------------------------------------------------------------------------
// fused node GRU: gi = agg @ Wih.T (K=256), gh = unf @ Whh.T (K=256),
// in-place update of unf (each block touches only its own 32 rows).
// ---------------------------------------------------------------------------
__global__ __launch_bounds__(256, 1)
void node_gru_kernel(float* __restrict__ unf, const float* __restrict__ agg,
                     const __bf16* __restrict__ Wih, const __bf16* __restrict__ Whh,
                     const float* __restrict__ bih, const float* __restrict__ bhh,
                     int M) {
  __shared__ float ag[32 * 256];
  __shared__ float hh[32 * 256];

  const int t  = threadIdx.x;
  const int n0 = blockIdx.x * 32;

#if HAVE_ASYNC_LDS
  for (int idx = t; idx < 2048; idx += 256) {
    int row = idx >> 6;
    int gr  = n0 + row; if (gr >= M) gr = M - 1;
    async_cp16((const f32x4*)agg + (size_t)gr * 64 + (idx & 63), (f32x4*)ag + idx);
    async_cp16((const f32x4*)unf + (size_t)gr * 64 + (idx & 63), (f32x4*)hh + idx);
  }
  async_wait0();
#else
  for (int idx = t; idx < 2048; idx += 256) {
    int row = idx >> 6;                 // 64 float4 per row
    int gr  = n0 + row; if (gr >= M) gr = M - 1;
    ((f32x4*)ag)[idx] = ((const f32x4*)agg)[(size_t)gr * 64 + (idx & 63)];
    ((f32x4*)hh)[idx] = ((const f32x4*)unf)[(size_t)gr * 64 + (idx & 63)];
  }
#endif
  __syncthreads();

  const int wave = t >> 5, lane = t & 31;

  for (int cb = wave; cb < 16; cb += 8) {
    const int c0 = cb * 16;
    f32x8 Sr[2] = {}, Sz[2] = {}, Gin[2] = {}, Ghn[2] = {};

#pragma unroll 4
    for (int ks = 0; ks < 8; ++ks) {
      const int kk = ks * 32;
      bf16x16 br = load_b_frag(Wih, 256, c0,       kk, lane);
      bf16x16 bz = load_b_frag(Wih, 256, 256 + c0, kk, lane);
      bf16x16 bn = load_b_frag(Wih, 256, 512 + c0, kk, lane);
#pragma unroll
      for (int s = 0; s < 2; ++s) {
        bf16x16 a = load_a_frag_f32(ag, 256, s * 16, kk, lane);
        Sr[s]  = wmma_bf16(a, br, Sr[s]);
        Sz[s]  = wmma_bf16(a, bz, Sz[s]);
        Gin[s] = wmma_bf16(a, bn, Gin[s]);
      }
    }
#pragma unroll 4
    for (int ks = 0; ks < 8; ++ks) {
      const int kk = ks * 32;
      bf16x16 br = load_b_frag(Whh, 256, c0,       kk, lane);
      bf16x16 bz = load_b_frag(Whh, 256, 256 + c0, kk, lane);
      bf16x16 bn = load_b_frag(Whh, 256, 512 + c0, kk, lane);
#pragma unroll
      for (int s = 0; s < 2; ++s) {
        bf16x16 a = load_a_frag_f32(hh, 256, s * 16, kk, lane);
        Sr[s]  = wmma_bf16(a, br, Sr[s]);
        Sz[s]  = wmma_bf16(a, bz, Sz[s]);
        Ghn[s] = wmma_bf16(a, bn, Ghn[s]);
      }
    }

    const int   c    = c0 + (lane & 15);
    const float br_  = bih[c]       + bhh[c];
    const float bz_  = bih[256 + c] + bhh[256 + c];
    const float bin_ = bih[512 + c];
    const float bhn_ = bhh[512 + c];
    const int   rtop = (lane >> 4) * 8;

#pragma unroll
    for (int s = 0; s < 2; ++s)
#pragma unroll
      for (int r = 0; r < 8; ++r) {
        const int nrow = s * 16 + rtop + r;
        if (n0 + nrow < M) {
          float rg = sigmoidf_(Sr[s][r] + br_);
          float zg = sigmoidf_(Sz[s][r] + bz_);
          float ng = tanhf(Gin[s][r] + bin_ + rg * (Ghn[s][r] + bhn_));
          float h  = hh[nrow * 256 + c];
          unf[(size_t)(n0 + nrow) * 256 + c] = (1.0f - zg) * ng + zg * h;
        }
      }
  }
}

// ---------------------------------------------------------------------------
// generic GEMM with bias: C[M,Nout] = A[M,K] @ W[Nout,K]^T + bias
// (encoders / decoders). K in {64,128,256}, Nout in {64,128,256}.
// ---------------------------------------------------------------------------
__global__ __launch_bounds__(256, 1)
void gemm_bias_kernel(const float* __restrict__ A, int K,
                      const __bf16* __restrict__ W, const float* __restrict__ bias,
                      float* __restrict__ C, int Nout, int M) {
  __shared__ float As[32 * 256];

  const int t  = threadIdx.x;
  const int m0 = blockIdx.x * 32;
  const int K4 = K >> 2;

#if HAVE_ASYNC_LDS
  for (int idx = t; idx < 32 * K4; idx += 256) {
    int row = idx / K4, col = idx - row * K4;
    int gr  = m0 + row; if (gr >= M) gr = M - 1;
    async_cp16((const f32x4*)A + (size_t)gr * K4 + col, (f32x4*)As + row * K4 + col);
  }
  async_wait0();
#else
  for (int idx = t; idx < 32 * K4; idx += 256) {
    int row = idx / K4, col = idx - row * K4;
    int gr  = m0 + row; if (gr >= M) gr = M - 1;
    ((f32x4*)As)[row * K4 + col] = ((const f32x4*)A)[(size_t)gr * K4 + col];
  }
#endif
  __syncthreads();

  const int wave = t >> 5, lane = t & 31;
  const int nks  = K >> 5;

  for (int cb = wave; cb * 16 < Nout; cb += 8) {
    const int c0 = cb * 16;
    f32x8 acc[2] = {};
    for (int ks = 0; ks < nks; ++ks) {
      const int kk = ks * 32;
      bf16x16 b = load_b_frag(W, K, c0, kk, lane);
#pragma unroll
      for (int s = 0; s < 2; ++s) {
        bf16x16 a = load_a_frag_f32(As, K, s * 16, kk, lane);
        acc[s] = wmma_bf16(a, b, acc[s]);
      }
    }
    const int   c    = c0 + (lane & 15);
    const float bv   = bias[c];
    const int   rtop = (lane >> 4) * 8;
#pragma unroll
    for (int s = 0; s < 2; ++s)
#pragma unroll
      for (int r = 0; r < 8; ++r) {
        const int row = m0 + s * 16 + rtop + r;
        if (row < M) C[(size_t)row * Nout + c] = acc[s][r] + bv;
      }
  }
}

// ---------------------------------------------------------------------------
__global__ void cvt_f32_bf16_kernel(const float* __restrict__ s,
                                    __bf16* __restrict__ d, int n) {
  int i = blockIdx.x * 256 + threadIdx.x;
  if (i < n) d[i] = (__bf16)s[i];
}

__global__ void zero_f32_kernel(float* __restrict__ p, int n) {
  int i = blockIdx.x * 256 + threadIdx.x;
  if (i < n) p[i] = 0.0f;
}

// ---------------------------------------------------------------------------
static inline void* ws_take(char*& p, size_t bytes) {
  void* r = (void*)p;
  p += (bytes + 255) & ~(size_t)255;
  return r;
}

extern "C" void kernel_launch(void* const* d_in, const int* in_sizes, int n_in,
                              void* d_out, int out_size, void* d_ws, size_t ws_size,
                              hipStream_t stream) {
  (void)in_sizes; (void)n_in; (void)out_size; (void)ws_size;

  const float* nf    = (const float*)d_in[0];
  const float* ef    = (const float*)d_in[1];
  const int*   src   = (const int*)d_in[2];
  const int*   dst   = (const int*)d_in[3];
  const float* Wne   = (const float*)d_in[4];
  const float* bne   = (const float*)d_in[5];
  const float* Wee   = (const float*)d_in[6];
  const float* bee   = (const float*)d_in[7];
  const float* Wih_e = (const float*)d_in[8];
  const float* Whh_e = (const float*)d_in[9];
  const float* bih_e = (const float*)d_in[10];
  const float* bhh_e = (const float*)d_in[11];
  const float* Wih_n = (const float*)d_in[12];
  const float* Whh_n = (const float*)d_in[13];
  const float* bih_n = (const float*)d_in[14];
  const float* bhh_n = (const float*)d_in[15];
  const float* Wnd   = (const float*)d_in[16];
  const float* bnd   = (const float*)d_in[17];
  const float* Wed   = (const float*)d_in[18];
  const float* bed   = (const float*)d_in[19];

  char* wp = (char*)d_ws;
  float*  uef     = (float*)ws_take(wp, (size_t)NEDGE * HID * 4);
  float*  unf     = (float*)ws_take(wp, (size_t)NNODE * HID * 4);
  float*  agg     = (float*)ws_take(wp, (size_t)NNODE * HID * 4);
  __bf16* unf_bf  = (__bf16*)ws_take(wp, (size_t)NNODE * HID * 2);
  __bf16* Wne_b   = (__bf16*)ws_take(wp, (size_t)256 * 128 * 2);
  __bf16* Wee_b   = (__bf16*)ws_take(wp, (size_t)256 * 64 * 2);
  __bf16* Wih_e_b = (__bf16*)ws_take(wp, (size_t)768 * 512 * 2);
  __bf16* Whh_e_b = (__bf16*)ws_take(wp, (size_t)768 * 256 * 2);
  __bf16* Wih_n_b = (__bf16*)ws_take(wp, (size_t)768 * 256 * 2);
  __bf16* Whh_n_b = (__bf16*)ws_take(wp, (size_t)768 * 256 * 2);
  __bf16* Wnd_b   = (__bf16*)ws_take(wp, (size_t)128 * 256 * 2);
  __bf16* Wed_b   = (__bf16*)ws_take(wp, (size_t)64 * 256 * 2);

  auto cvt = [&](const float* s, __bf16* d, int n) {
    cvt_f32_bf16_kernel<<<(n + 255) / 256, 256, 0, stream>>>(s, d, n);
  };

  // 1) bf16 weight shadows (tiny; live in L2 thereafter)
  cvt(Wne,   Wne_b,   256 * 128);
  cvt(Wee,   Wee_b,   256 * 64);
  cvt(Wih_e, Wih_e_b, 768 * 512);
  cvt(Whh_e, Whh_e_b, 768 * 256);
  cvt(Wih_n, Wih_n_b, 768 * 256);
  cvt(Whh_n, Whh_n_b, 768 * 256);
  cvt(Wnd,   Wnd_b,   128 * 256);
  cvt(Wed,   Wed_b,   64 * 256);

  const int nodeBlocks = (NNODE + 31) / 32;   // 313
  const int edgeBlocks = NEDGE / 32;          // 10000

  // 2) encoders
  gemm_bias_kernel<<<nodeBlocks, 256, 0, stream>>>(nf, 128, Wne_b, bne, unf, 256, NNODE);
  gemm_bias_kernel<<<edgeBlocks, 256, 0, stream>>>(ef,  64, Wee_b, bee, uef, 256, NEDGE);
  cvt(unf, unf_bf, NNODE * HID);

  // 3) message passing
  for (int it = 0; it < 3; ++it) {
    zero_f32_kernel<<<(NNODE * HID + 255) / 256, 256, 0, stream>>>(agg, NNODE * HID);
    edge_gru_kernel<<<edgeBlocks, 256, 0, stream>>>(unf_bf, uef, src, dst,
                                                    Wih_e_b, Whh_e_b, bih_e, bhh_e, agg);
    node_gru_kernel<<<nodeBlocks, 256, 0, stream>>>(unf, agg, Wih_n_b, Whh_n_b,
                                                    bih_n, bhh_n, NNODE);
    cvt(unf, unf_bf, NNODE * HID);
  }

  // 4) decoders -> d_out = [node_out (N x 128) | edge_out (E x 64)]
  float* out_n = (float*)d_out;
  float* out_e = out_n + (size_t)NNODE * 128;
  gemm_bias_kernel<<<nodeBlocks, 256, 0, stream>>>(unf, 256, Wnd_b, bnd, out_n, 128, NNODE);
  gemm_bias_kernel<<<edgeBlocks, 256, 0, stream>>>(uef, 256, Wed_b, bed, out_e, 64, NEDGE);
}
